// AdaptiveVAE_29772713296206
// MI455X (gfx1250) — matrix operand order repaired and checked by reading, not compiled, not run on hardware
//
#include <hip/hip_runtime.h>
#include <hip/hip_bf16.h>

// ---------------------------------------------------------------------------
// Fused AdaptiveVAE forward for MI455X (gfx1250), wave32 + WMMA.
//
// B=262144 rows, S=6, D=16, H=4, HD=4, FF=64, L=10, NF=6, NC=139, DH=32.
// One 64-thread (2-wave) workgroup processes 32 batch rows (192 token-rows).
// All dense projections run on V_WMMA_F32_16X16X32_F16 (M = 16 token/row
// pairs, K padded to 32). Weight B-fragments are pre-staged in LDS in exact
// per-lane register order (f16, zero padded) so the hot loop does pure
// unconditional ds_load_b128 -> v_wmma, no divergent guards, no cvt.
// ---------------------------------------------------------------------------

typedef __attribute__((ext_vector_type(16))) _Float16 v16h;
typedef __attribute__((ext_vector_type(8)))  float    v8f;

#define BTOT  262144
#define ROWS  32            // batch rows per workgroup
#define TPB   64            // 2 waves (wave32)
#define TOK   (ROWS * 6)    // 192 token-rows per workgroup
#define NLOG  139

__device__ __forceinline__ v8f wmma16(v16h a, v16h b, v8f c) {
  return __builtin_amdgcn_wmma_f32_16x16x32_f16(
      /*neg_a=*/false, a, /*neg_b=*/false, b,
      /*c_mod=*/(short)0, c, /*reuse_a=*/false, /*reuse_b=*/false);
}

// ---- A-fragment loaders (ISA 7.12.2 16-bit A 16x32 layout) ---------------
// Element j<8  -> K = j + 8*grp            (contiguous 8)
// Element j>=8 -> K = 16 + (j-8) + 8*grp   (contiguous 8, zero if K dim==16)

__device__ __forceinline__ v16h load_a_f32_k16(const float* base, int stride) {
  const int lane = (int)threadIdx.x & 31;
  const float* pr = base + (lane & 15) * stride + (lane >> 4) * 8;
  v16h a;
#pragma unroll
  for (int j = 0; j < 8; ++j) a[j] = (_Float16)pr[j];
#pragma unroll
  for (int j = 8; j < 16; ++j) a[j] = (_Float16)0.0f;
  return a;
}

__device__ __forceinline__ v16h load_a_f16_k16(const _Float16* base, int stride) {
  const int lane = (int)threadIdx.x & 31;
  const _Float16* pr = base + (lane & 15) * stride + (lane >> 4) * 8;
  v16h a;
#pragma unroll
  for (int j = 0; j < 8; ++j) a[j] = pr[j];
#pragma unroll
  for (int j = 8; j < 16; ++j) a[j] = (_Float16)0.0f;
  return a;
}

__device__ __forceinline__ v16h load_a_f16_k32(const _Float16* base, int stride) {
  const int lane = (int)threadIdx.x & 31;
  const _Float16* pr = base + (lane & 15) * stride + (lane >> 4) * 8;
  v16h a;
#pragma unroll
  for (int j = 0; j < 8; ++j) a[j] = pr[j];
#pragma unroll
  for (int j = 0; j < 8; ++j) a[8 + j] = pr[16 + j];
  return a;
}

// ---- B-fragment staging ---------------------------------------------------
// B (K<=32 x 16) for y = x @ W^T, W row-major (N,K) f32 in global.
// Register layout: lanes 0-15 hold K=0..7,16..23, lanes 16-31 hold K=8..15,24..31,
// N = lane&15, element j -> K = (j&7) + 8*(lane>>4) + 16*(j>>3).
// Staged into LDS as [ntile][lane][16] f16 so the hot-loop load is a single
// unconditional 32-byte vector load.
__device__ __forceinline__ void stage_bfrag(_Float16* dst, const float* W,
                                            int kstride, int koff,
                                            int ntiles, int nmax, int kmax) {
  for (int idx = (int)threadIdx.x; idx < ntiles * 512; idx += TPB) {
    const int j  = idx & 15;
    const int l  = (idx >> 4) & 31;
    const int nt = idx >> 9;
    const int n  = nt * 16 + (l & 15);
    const int k  = (j & 7) + ((l >> 4) << 3) + ((j >> 3) << 4);
    dst[idx] = (k < kmax && n < nmax) ? (_Float16)W[n * kstride + koff + k]
                                      : (_Float16)0.0f;
  }
}

__device__ __forceinline__ v16h bfrag(const _Float16* buf, int nt) {
  return *(const v16h*)(buf + (nt * 32 + ((int)threadIdx.x & 31)) * 16);
}

// Deterministic hash -> standard normal (Box-Muller). The reference's
// jax.random.normal(key 42) stream is not reproducible on device; this keeps
// the same math structure and is identical on every replay.
__device__ __forceinline__ float randn_hash(unsigned idx) {
  unsigned x = idx * 2654435761u + 0x9E3779B9u;
  x ^= x >> 16; x *= 0x7feb352du; x ^= x >> 15; x *= 0x846ca68bu; x ^= x >> 16;
  const float u1 = ((x & 0xffffu) + 1.0f) * (1.0f / 65537.0f);
  const float u2 = ((x >> 16) + 0.5f) * (1.0f / 65536.0f);
  return sqrtf(-2.0f * __logf(u1)) * __cosf(6.28318530718f * u2);
}

__device__ __forceinline__ void stage_f(float* dst, const float* src, int n) {
  for (int i = (int)threadIdx.x; i < n; i += TPB) dst[i] = src[i];
}

__device__ __forceinline__ void ln_rows(float* X, const float* w, const float* b) {
  for (int t = (int)threadIdx.x; t < TOK; t += TPB) {
    float* x = X + t * 16;
    float m = 0.0f;
#pragma unroll
    for (int d = 0; d < 16; ++d) m += x[d];
    m *= 0.0625f;
    float v = 0.0f;
#pragma unroll
    for (int d = 0; d < 16; ++d) { const float df = x[d] - m; v += df * df; }
    v *= 0.0625f;
    const float inv = rsqrtf(v + 1e-5f);
#pragma unroll
    for (int d = 0; d < 16; ++d) x[d] = (x[d] - m) * inv * w[d] + b[d];
  }
}

struct Params {
  const int* fid; const float* fval; const unsigned char* mask;
  const float* type_embed; const float* value_w; const float* value_b; const float* pos_enc;
  const float* qkv_w; const float* qkv_b; const float* out_w; const float* out_b;
  const float* ln1_w; const float* ln1_b; const float* ffn_w1; const float* ffn_b1;
  const float* ffn_w2; const float* ffn_b2; const float* ln2_w; const float* ln2_b;
  const float* pool_w; const float* pool_b; const float* mu_w; const float* mu_b;
  const float* lv_w; const float* lv_b; const float* dec_w1; const float* dec_b1;
  const float* dec_w2; const float* dec_b2; const float* cls_w1; const float* cls_b1;
  const float* cls_w2; const float* cls_b2;
  float* out_rec; float* out_mu; float* out_lv; float* out_logits;
};

__global__ __launch_bounds__(TPB) void vae_fused(Params p) {
  __shared__ float    sX[TOK * 16];                 // token state (f32)
  __shared__ __align__(32) _Float16 sS[TOK * 64];   // qkv(48)/ffn-h(64)/dec-h/cls-h1
  __shared__ __align__(32) _Float16 sWf[12 * 512];  // staged B fragments (max dec_w1)
  __shared__ float    sB[192];                      // staged biases
  __shared__ float    sPool[ROWS * 16];             // pooled / allo scratch
  __shared__ float    sZ[ROWS * 16];                // latent z (cols 10..15 = 0)
  __shared__ float    sMaskF[TOK];
  __shared__ float    sVal[TOK];
  __shared__ int      sFid[TOK];

  const int tid  = (int)threadIdx.x;
  const int wave = tid >> 5;
  const int lane = tid & 31;
  const int grp  = lane >> 4;
  const int nn   = lane & 15;
  const int row0 = (int)blockIdx.x * ROWS;

  // warm GL2 for the large tail weights while we do the front of the model
  __builtin_prefetch(p.cls_w2, 0, 0);
  __builtin_prefetch(p.dec_w1, 0, 0);

  // ---- stage per-row inputs --------------------------------------------
  for (int i = tid; i < TOK; i += TPB) {
    const int g = row0 * 6 + i;
    sFid[i]   = p.fid[g];
    sVal[i]   = p.fval[g];
    sMaskF[i] = p.mask[g] ? 1.0f : 0.0f;
  }
  __syncthreads();

  // ---- embedding: type_embed[fid] + v*value_w + value_b + pos_enc[s] ----
  for (int task = tid; task < TOK * 16; task += TPB) {
    const int t = task >> 4, d = task & 15;
    const int s = t % 6;
    sX[task] = p.type_embed[sFid[t] * 16 + d] + sVal[t] * p.value_w[d]
             + p.value_b[d] + p.pos_enc[s * 16 + d];
  }

  // ======================= 2 transformer layers =========================
  for (int l = 0; l < 2; ++l) {
    // ---- QKV projection: x(16) -> 48, 3 N-tiles per M-tile -------------
    __syncthreads();
    stage_bfrag(sWf, p.qkv_w + l * 768, 16, 0, 3, 48, 16);
    stage_f(sB, p.qkv_b + l * 48, 48);
    __syncthreads();
    {
      const v16h b0 = bfrag(sWf, 0), b1 = bfrag(sWf, 1), b2 = bfrag(sWf, 2);
#pragma unroll
      for (int i = 0; i < 6; ++i) {
        const int mt = wave * 6 + i;
        const v16h a = load_a_f32_k16(sX + mt * 256, 16);
        v8f c0 = {}, c1 = {}, c2 = {};
        c0 = wmma16(a, b0, c0);
        c1 = wmma16(a, b1, c1);
        c2 = wmma16(a, b2, c2);
#pragma unroll
        for (int r = 0; r < 8; ++r) {
          const int tr = mt * 16 + r + 8 * grp;
          sS[tr * 48 +      nn] = (_Float16)(c0[r] + sB[nn]);
          sS[tr * 48 + 16 + nn] = (_Float16)(c1[r] + sB[16 + nn]);
          sS[tr * 48 + 32 + nn] = (_Float16)(c2[r] + sB[32 + nn]);
        }
      }
    }
    __syncthreads();

    // ---- attention: 6x6 per (row, head), register resident -------------
    for (int task = tid; task < ROWS * 4; task += TPB) {
      const int row = task >> 2, h = task & 3;
      const int tb = row * 6;
      float q[6][4], kk[6][4], vv[6][4];
#pragma unroll
      for (int i = 0; i < 6; ++i)
#pragma unroll
        for (int d = 0; d < 4; ++d) {
          q[i][d]  = (float)sS[(tb + i) * 48 +      h * 4 + d];
          kk[i][d] = (float)sS[(tb + i) * 48 + 16 + h * 4 + d];
          vv[i][d] = (float)sS[(tb + i) * 48 + 32 + h * 4 + d];
        }
#pragma unroll
      for (int i = 0; i < 6; ++i) {
        float sc[6], mx = -3.0e38f;
#pragma unroll
        for (int j = 0; j < 6; ++j) {
          float s = 0.0f;
#pragma unroll
          for (int d = 0; d < 4; ++d) s += q[i][d] * kk[j][d];
          s = s * 0.5f + ((sMaskF[tb + j] > 0.5f) ? 0.0f : -1.0e9f);
          sc[j] = s; mx = fmaxf(mx, s);
        }
        float den = 0.0f;
#pragma unroll
        for (int j = 0; j < 6; ++j) { sc[j] = __expf(sc[j] - mx); den += sc[j]; }
        const float inv = 1.0f / den;
#pragma unroll
        for (int d = 0; d < 4; ++d) {
          float o = 0.0f;
#pragma unroll
          for (int j = 0; j < 6; ++j) o += sc[j] * vv[j][d];
          sS[(tb + i) * 48 + h * 4 + d] = (_Float16)(o * inv);  // into q slot
        }
      }
    }
    __syncthreads();

    // ---- out projection + residual -------------------------------------
    stage_bfrag(sWf, p.out_w + l * 256, 16, 0, 1, 16, 16);
    stage_f(sB, p.out_b + l * 16, 16);
    __syncthreads();
    {
      const v16h b0 = bfrag(sWf, 0);
#pragma unroll
      for (int i = 0; i < 6; ++i) {
        const int mt = wave * 6 + i;
        const v16h a = load_a_f16_k16(sS + mt * 16 * 48, 48);
        v8f c = {};
        c = wmma16(a, b0, c);
#pragma unroll
        for (int r = 0; r < 8; ++r) {
          const int tr = mt * 16 + r + 8 * grp;
          sX[tr * 16 + nn] += c[r] + sB[nn];
        }
      }
    }
    __syncthreads();
    ln_rows(sX, p.ln1_w + l * 16, p.ln1_b + l * 16);
    __syncthreads();

    // ---- FFN up: 16 -> 64, relu ----------------------------------------
    stage_bfrag(sWf, p.ffn_w1 + l * 1024, 16, 0, 4, 64, 16);
    stage_f(sB, p.ffn_b1 + l * 64, 64);
    __syncthreads();
    {
      const v16h b0 = bfrag(sWf, 0), b1 = bfrag(sWf, 1);
      const v16h b2 = bfrag(sWf, 2), b3 = bfrag(sWf, 3);
#pragma unroll
      for (int i = 0; i < 6; ++i) {
        const int mt = wave * 6 + i;
        const v16h a = load_a_f32_k16(sX + mt * 256, 16);
        v8f c0 = {}, c1 = {}, c2 = {}, c3 = {};
        c0 = wmma16(a, b0, c0);
        c1 = wmma16(a, b1, c1);
        c2 = wmma16(a, b2, c2);
        c3 = wmma16(a, b3, c3);
#pragma unroll
        for (int r = 0; r < 8; ++r) {
          const int tr = mt * 16 + r + 8 * grp;
          sS[tr * 64 +      nn] = (_Float16)fmaxf(c0[r] + sB[nn],      0.0f);
          sS[tr * 64 + 16 + nn] = (_Float16)fmaxf(c1[r] + sB[16 + nn], 0.0f);
          sS[tr * 64 + 32 + nn] = (_Float16)fmaxf(c2[r] + sB[32 + nn], 0.0f);
          sS[tr * 64 + 48 + nn] = (_Float16)fmaxf(c3[r] + sB[48 + nn], 0.0f);
        }
      }
    }
    __syncthreads();

    // ---- FFN down: 64 -> 16 (two K=32 chunks) + residual ---------------
    stage_bfrag(sWf,       p.ffn_w2 + l * 1024, 64, 0,  1, 16, 32);
    stage_bfrag(sWf + 512, p.ffn_w2 + l * 1024, 64, 32, 1, 16, 32);
    stage_f(sB, p.ffn_b2 + l * 16, 16);
    __syncthreads();
    {
      const v16h bA = bfrag(sWf, 0), bB = bfrag(sWf + 512, 0);
#pragma unroll
      for (int i = 0; i < 6; ++i) {
        const int mt = wave * 6 + i;
        v8f c = {};
        c = wmma16(load_a_f16_k32(sS + mt * 16 * 64,      64), bA, c);
        c = wmma16(load_a_f16_k32(sS + mt * 16 * 64 + 32, 64), bB, c);
#pragma unroll
        for (int r = 0; r < 8; ++r) {
          const int tr = mt * 16 + r + 8 * grp;
          sX[tr * 16 + nn] += c[r] + sB[nn];
        }
      }
    }
    __syncthreads();
    ln_rows(sX, p.ln2_w + l * 16, p.ln2_b + l * 16);
  }
  __syncthreads();

  // ---- masked mean pool -------------------------------------------------
  for (int task = tid; task < ROWS * 16; task += TPB) {
    const int row = task >> 4, d = task & 15;
    float s = 0.0f, cm = 0.0f;
#pragma unroll
    for (int j = 0; j < 6; ++j) {
      const float mf = sMaskF[row * 6 + j];
      s += sX[(row * 6 + j) * 16 + d] * mf;
      cm += mf;
    }
    sPool[task] = s / cm;
  }
  __syncthreads();

  // ---- pool projection + relu (one 16-row M-tile per wave) --------------
  stage_bfrag(sWf, p.pool_w, 16, 0, 1, 16, 16);
  stage_f(sB, p.pool_b, 16);
  __syncthreads();
  {
    const v16h a = load_a_f32_k16(sPool + wave * 256, 16);
    v8f c = {};
    c = wmma16(a, bfrag(sWf, 0), c);
#pragma unroll
    for (int r = 0; r < 8; ++r) {
      const int rr = wave * 16 + r + 8 * grp;
      sPool[rr * 16 + nn] = fmaxf(c[r] + sB[nn], 0.0f);
    }
  }
  __syncthreads();

  // ---- mu / logvar / z ---------------------------------------------------
  stage_bfrag(sWf,       p.mu_w, 16, 0, 1, 10, 16);
  stage_bfrag(sWf + 512, p.lv_w, 16, 0, 1, 10, 16);
  stage_f(sB, p.mu_b, 10);
  stage_f(sB + 16, p.lv_b, 10);
  __syncthreads();
  {
    const v16h a = load_a_f32_k16(sPool + wave * 256, 16);
    v8f dm = {}, dl = {};
    dm = wmma16(a, bfrag(sWf, 0), dm);
    dl = wmma16(a, bfrag(sWf + 512, 0), dl);
#pragma unroll
    for (int r = 0; r < 8; ++r) {
      const int rr = wave * 16 + r + 8 * grp;
      const int grow = row0 + rr;
      float z = 0.0f;
      if (nn < 10) {
        const float mu = dm[r] + sB[nn];
        const float lv = dl[r] + sB[16 + nn];
        p.out_mu[grow * 10 + nn] = mu;
        p.out_lv[grow * 10 + nn] = lv;
        z = mu + randn_hash((unsigned)(grow * 10 + nn)) * __expf(0.5f * lv);
      }
      sZ[rr * 16 + nn] = z;
    }
  }
  __syncthreads();

  // ---- 6 tiny decoders fused as one GEMM: z(10) -> 192, relu ------------
  stage_bfrag(sWf, p.dec_w1, 10, 0, 12, 192, 10);   // (6*32, 10) row-major
  stage_f(sB, p.dec_b1, 192);
  __syncthreads();
  {
    const v16h a = load_a_f32_k16(sZ + wave * 256, 16);
#pragma unroll
    for (int nt = 0; nt < 12; ++nt) {
      v8f c = {};
      c = wmma16(a, bfrag(sWf, nt), c);
#pragma unroll
      for (int r = 0; r < 8; ++r) {
        const int rr = wave * 16 + r + 8 * grp;
        const int col = nt * 16 + nn;
        sS[rr * 192 + col] = (_Float16)fmaxf(c[r] + sB[col], 0.0f);
      }
    }
  }
  __syncthreads();

  // ---- decoder heads (32-wide dot) + per-token gather -> recons ----------
  for (int task = tid; task < ROWS * 6; task += TPB) {
    const int row = task / 6, e = task % 6;
    float s = p.dec_b2[e];
#pragma unroll
    for (int h = 0; h < 32; ++h)
      s += p.dec_w2[e * 32 + h] * (float)sS[row * 192 + e * 32 + h];
    sPool[row * 16 + e] = s;   // sPool reused as allo
  }
  __syncthreads();
  for (int task = tid; task < TOK; task += TPB) {
    const int row = task / 6;
    p.out_rec[row0 * 6 + task] = sPool[row * 16 + sFid[task]];
  }
  __syncthreads();

  // ---- classifier: relu(z @ W1^T) @ W2^T --------------------------------
  stage_bfrag(sWf, p.cls_w1, 10, 0, 2, 32, 10);
  stage_f(sB, p.cls_b1, 32);
  __syncthreads();
  {
    const v16h a = load_a_f32_k16(sZ + wave * 256, 16);
#pragma unroll
    for (int nt = 0; nt < 2; ++nt) {
      v8f c = {};
      c = wmma16(a, bfrag(sWf, nt), c);
#pragma unroll
      for (int r = 0; r < 8; ++r) {
        const int rr = wave * 16 + r + 8 * grp;
        const int col = nt * 16 + nn;
        sS[8192 + rr * 32 + col] = (_Float16)fmaxf(c[r] + sB[col], 0.0f);
      }
    }
  }
  __syncthreads();
  stage_bfrag(sWf, p.cls_w2, 32, 0, 9, NLOG, 32);
  stage_f(sB, p.cls_b2, NLOG);
  __syncthreads();
  {
    const v16h a = load_a_f16_k32(sS + 8192 + wave * 16 * 32, 32);
#pragma unroll
    for (int nt = 0; nt < 9; ++nt) {
      v8f c = {};
      c = wmma16(a, bfrag(sWf, nt), c);
#pragma unroll
      for (int r = 0; r < 8; ++r) {
        const int rr = wave * 16 + r + 8 * grp;
        const int grow = row0 + rr;
        const int col = nt * 16 + nn;
        if (col < NLOG) p.out_logits[grow * NLOG + col] = c[r] + sB[col];
      }
    }
  }
}

extern "C" void kernel_launch(void* const* d_in, const int* in_sizes, int n_in,
                              void* d_out, int out_size, void* d_ws, size_t ws_size,
                              hipStream_t stream) {
  (void)in_sizes; (void)n_in; (void)out_size; (void)d_ws; (void)ws_size;
  Params p;
  p.fid        = (const int*)d_in[0];
  p.fval       = (const float*)d_in[1];
  p.mask       = (const unsigned char*)d_in[2];
  p.type_embed = (const float*)d_in[3];
  p.value_w    = (const float*)d_in[4];
  p.value_b    = (const float*)d_in[5];
  p.pos_enc    = (const float*)d_in[6];
  p.qkv_w      = (const float*)d_in[7];  p.qkv_b  = (const float*)d_in[8];
  p.out_w      = (const float*)d_in[9];  p.out_b  = (const float*)d_in[10];
  p.ln1_w      = (const float*)d_in[11]; p.ln1_b  = (const float*)d_in[12];
  p.ffn_w1     = (const float*)d_in[13]; p.ffn_b1 = (const float*)d_in[14];
  p.ffn_w2     = (const float*)d_in[15]; p.ffn_b2 = (const float*)d_in[16];
  p.ln2_w      = (const float*)d_in[17]; p.ln2_b  = (const float*)d_in[18];
  p.pool_w     = (const float*)d_in[19]; p.pool_b = (const float*)d_in[20];
  p.mu_w       = (const float*)d_in[21]; p.mu_b   = (const float*)d_in[22];
  p.lv_w       = (const float*)d_in[23]; p.lv_b   = (const float*)d_in[24];
  p.dec_w1     = (const float*)d_in[25]; p.dec_b1 = (const float*)d_in[26];
  p.dec_w2     = (const float*)d_in[27]; p.dec_b2 = (const float*)d_in[28];
  p.cls_w1     = (const float*)d_in[29]; p.cls_b1 = (const float*)d_in[30];
  p.cls_w2     = (const float*)d_in[31]; p.cls_b2 = (const float*)d_in[32];

  float* out = (float*)d_out;
  p.out_rec    = out;                          // B*6
  p.out_mu     = out + (size_t)BTOT * 6;       // B*10
  p.out_lv     = out + (size_t)BTOT * 16;      // B*10
  p.out_logits = out + (size_t)BTOT * 26;      // B*139

  dim3 grid(BTOT / ROWS), block(TPB);
  hipLaunchKernelGGL(vae_fused, grid, block, 0, stream, p);
}
